// DCGRUCell_52707838656930
// MI455X (gfx1250) — matrix operand order repaired and checked by reading, not compiled, not run on hardware
//
#include <hip/hip_runtime.h>
#include <math.h>

// Problem constants (match reference)
#define NN   4096          // nodes
#define BB   32            // batch
#define UU   64            // units
#define FF   66            // IN_DIM + U
#define MMAT 5             // num diffusion matrices
#define CC   (FF * BB)     // 2112 columns of X
#define KTOT 330           // F * M
#define KPAD 352           // 11 * 32 (K padded for WMMA)

typedef __attribute__((ext_vector_type(16))) __bf16 bf16x16;
typedef __attribute__((ext_vector_type(4)))  __bf16 bf16x4;
typedef __attribute__((ext_vector_type(8)))  float  f32x8;
typedef __attribute__((ext_vector_type(4)))  unsigned int u32x4;
typedef __attribute__((ext_vector_type(8)))  unsigned int u32x8;

__device__ __forceinline__ __bf16 f2bf(float f) {
  union { float f; unsigned u; } v; v.f = f;
  unsigned r = v.u + 0x7FFFu + ((v.u >> 16) & 1u);   // round-to-nearest-even
  unsigned short h = (unsigned short)(r >> 16);
  return __builtin_bit_cast(__bf16, h);
}
__device__ __forceinline__ float bf2f(__bf16 b) {
  unsigned short h = __builtin_bit_cast(unsigned short, b);
  union { unsigned u; float f; } v; v.u = ((unsigned)h) << 16;
  return v.f;
}

// ---------------------------------------------------------------- utilities
__global__ __launch_bounds__(256) void k_f32_to_bf16(
    const float* __restrict__ src, __bf16* __restrict__ dst, int n) {
  int i = blockIdx.x * 256 + threadIdx.x;
  if (i < n) dst[i] = f2bf(src[i]);
}

// Wt[o*KPAD + k] = W[k*out + o] (bf16, zero-padded K)
__global__ __launch_bounds__(256) void k_pack_w(
    const float* __restrict__ W, __bf16* __restrict__ Wt, int out) {
  int i = blockIdx.x * 256 + threadIdx.x;
  if (i >= out * KPAD) return;
  int o = i / KPAD, k = i - o * KPAD;
  Wt[i] = (k < KTOT) ? f2bf(W[k * out + o]) : f2bf(0.f);
}

// X0[n*CC + f*32 + b] = concat(inputs, state) in bf16
__global__ __launch_bounds__(256) void k_build_x(
    const float* __restrict__ inp, const float* __restrict__ st,
    __bf16* __restrict__ X0) {
  size_t i = (size_t)blockIdx.x * 256 + threadIdx.x;
  if (i >= (size_t)NN * CC) return;
  int n = (int)(i / CC);
  int r = (int)(i - (size_t)n * CC);
  int f = r >> 5, b = r & 31;
  float v = (f < 2) ? inp[(size_t)b * (NN * 2) + (size_t)n * 2 + f]
                    : st [(size_t)b * (NN * UU) + (size_t)n * UU + (f - 2)];
  X0[i] = f2bf(v);
}

// ------------------------------------------------- diffusion GEMM (bf16 WMMA)
// Y = alpha * (S @ X) + beta * Z     all (NN x CC), bf16, f32 accumulate.
// Block: 256 thr = 8 waves, 128x64 output tile, each wave a 32x32 patch.
// A tile (128x32 rows of S) is DMA'd into LDS by the Tensor Data Mover with
// hardware row padding (64B data + 16B pad -> LDA=40), double buffered on
// TENSORcnt. B tile is staged transposed by the threads (needs a transpose).
#define GT_M 128
#define GT_N 64
#define GT_K 32
#define LDA  40   // padded LDS row stride (bf16 units); TDM pad reproduces it
#define LDB  40

// 2D TDM descriptor: tile GT_K x GT_M (x-contig) from a NN x NN bf16 tensor.
// Group1 d0: data_size=1(2B)<<16 | pad_enable<<20 | pad_interval=3(64B)<<22
//            | pad_amount=3(16B)<<25
__device__ __forceinline__ void tdm_load_a_tile(const __bf16* gsrc,
                                                unsigned int lds_addr) {
  unsigned long long ga = (unsigned long long)(uintptr_t)gsrc;
  u32x4 g0;
  g0[0] = 1u;                                    // count=1, user mode
  g0[1] = lds_addr;                              // LDS byte address
  g0[2] = (unsigned int)ga;                      // global_addr[31:0]
  g0[3] = ((unsigned int)(ga >> 32) & 0x1FFFFFFu) | 0x80000000u; // [56:32]|type=2
  u32x8 g1;
  g1[0] = (1u << 16) | (1u << 20) | (3u << 22) | (3u << 25);
  g1[1] = ((unsigned int)(NN & 0xFFFF)) << 16;   // tensor_dim0[15:0] @ [63:48]
  g1[2] = ((NN >> 16) & 0xFFFF) | ((unsigned int)(NN & 0xFFFF) << 16); // d0 hi | d1 lo
  g1[3] = ((NN >> 16) & 0xFFFF) | ((unsigned int)GT_K << 16); // d1 hi | tile_dim0
  g1[4] = (unsigned int)GT_M;                    // tile_dim1 (tile_dim2 = 0)
  g1[5] = (unsigned int)NN;                      // tensor_dim0_stride[31:0]
  g1[6] = 0u;                                    // stride hi | dim1_stride lo
  g1[7] = 0u;
  asm volatile("tensor_load_to_lds %0, %1" :: "s"(g0), "s"(g1) : "memory");
}

template <bool HAS_BETA>
__global__ __launch_bounds__(256) void k_diff_gemm(
    const __bf16* __restrict__ S, const __bf16* __restrict__ X,
    const __bf16* __restrict__ Z, __bf16* __restrict__ Y,
    float alpha, float beta) {
  __shared__ __align__(16) __bf16 As[2][GT_M * LDA];  // TDM dest: As[m][k]
  __shared__ __align__(16) __bf16 Bs[GT_N * LDB];     // transposed: Bs[n][k]

  const int tid  = threadIdx.x;
  const int lane = tid & 31;
  const int wid  = tid >> 5;
  const int rowBase = blockIdx.x * GT_M;
  const int colBase = blockIdx.y * GT_N;
  const int wr = (wid & 3) * 32;   // wave row offset inside tile
  const int wc = (wid >> 2) * 32;  // wave col offset inside tile

  const unsigned int asBase =
      (unsigned int)(uintptr_t)(&As[0][0]);   // low 32 bits = LDS byte offset
  const unsigned int asStride = GT_M * LDA * 2;

  // B staging map: thread owns 4 consecutive cols nB..nB+3 at rows kB, kB+16.
  // -> two global_load_b64 at fixed immediate offsets off one stepping base,
  //    8 ds_store_b16 at fixed immediate offsets off one LDS base.
  const int nB = (tid & 15) * 4;
  const int kB = tid >> 4;                       // 0..15
  const __bf16* xp = X + (size_t)kB * CC + colBase + nB;
  __bf16* bsp = &Bs[nB * LDB + kB];

  f32x8 acc[2][2] = {};

  if (wid == 0)   // prologue: DMA first A tile
    tdm_load_a_tile(S + (size_t)rowBase * NN, asBase);

  for (int kt = 0; kt < NN; kt += GT_K) {
    const int cur = (kt >> 5) & 1;
    // Stage B: 32x64 bf16, transpose into Bs[n][k]
    bf16x4 v0 = *(const bf16x4*)(xp);                      // row kB
    bf16x4 v1 = *(const bf16x4*)(xp + (size_t)16 * CC);    // row kB+16
#pragma unroll
    for (int j = 0; j < 4; ++j) {
      bsp[j * LDB]      = v0[j];
      bsp[j * LDB + 16] = v1[j];
    }
    xp += (size_t)GT_K * CC;

    if (wid == 0) {
      if ((kt + GT_K) < NN) {      // DMA next A tile into the other buffer
        tdm_load_a_tile(S + (size_t)rowBase * NN + kt + GT_K,
                        asBase + (cur ^ 1) * asStride);
        // TDM ops are in-order: <=1 outstanding means tile `cur` landed.
        __builtin_amdgcn_s_wait_tensorcnt(1);
      } else {
        __builtin_amdgcn_s_wait_tensorcnt(0);
      }
    }
    __syncthreads();

    const __bf16* Ac = &As[cur][0];
    bf16x16 af[2], bfr[2];
    const int kh8  = (lane >> 4) << 3;  // A half-K select per ISA layout
    const int kh16 = (lane >> 4) << 4;  // unused; B also selects its half-K with kh8
#pragma unroll
    for (int s = 0; s < 2; ++s) {
      int m = wr + s * 16 + (lane & 15);
#pragma unroll
      for (int i = 0; i < 8; ++i) {
        af[s][i]     = Ac[m * LDA + kh8 + i];
        af[s][i + 8] = Ac[m * LDA + 16 + kh8 + i];
      }
      int n = wc + s * 16 + (lane & 15);
#pragma unroll
      for (int i = 0; i < 16; ++i) bfr[s][i] = Bs[n * LDB + (i >> 3) * 16 + kh8 + (i & 7)];
    }
#pragma unroll
    for (int sr = 0; sr < 2; ++sr)
#pragma unroll
      for (int sc = 0; sc < 2; ++sc)
        acc[sr][sc] = __builtin_amdgcn_wmma_f32_16x16x32_bf16(
            false, af[sr], false, bfr[sc], (short)0, acc[sr][sc], false, false);
    __syncthreads();
  }

  // Epilogue: C/D layout -> lane = N, vgpr r = M (+8 for upper lane half).
#pragma unroll
  for (int sr = 0; sr < 2; ++sr) {
#pragma unroll
    for (int sc = 0; sc < 2; ++sc) {
      const int gn = colBase + wc + sc * 16 + (lane & 15);
      const int gm0 = rowBase + wr + sr * 16 + ((lane >> 4) << 3);
      __bf16* yp = Y + (size_t)gm0 * CC + gn;
      const __bf16* zp = Z + (size_t)gm0 * CC + gn;
#pragma unroll
      for (int r = 0; r < 8; ++r) {
        float v = alpha * acc[sr][sc][r];
        if (HAS_BETA) v += beta * bf2f(zp[(size_t)r * CC]);
        yp[(size_t)r * CC] = f2bf(v);
      }
    }
  }
}

// ------------------------------------------------ projection GEMM + epilogue
// One block per node n. Computes (32 batches) x OUT = xs @ W + bias with WMMA.
// MODE 0: gate = sigmoid(.); cols<U -> rhx = gate*hx, cols>=U -> ubuf = gate
// MODE 1: c = tanh(.);  out = u*hx + (1-u)*c   (final new_state)
template <int OUT, int MODE>
__global__ __launch_bounds__(256) void k_proj(
    const __bf16* __restrict__ X0, const __bf16* __restrict__ X1,
    const __bf16* __restrict__ X2, const __bf16* __restrict__ X3,
    const __bf16* __restrict__ X4,
    const __bf16* __restrict__ Wt,    // (OUT, KPAD) bf16
    const float* __restrict__ bias,   // (OUT)
    const float* __restrict__ hx,     // (B, N*U)
    const float* __restrict__ ubuf,   // MODE 1 input
    float* __restrict__ out0,         // MODE0: rhx, MODE1: new_state
    float* __restrict__ out1) {       // MODE0: ubuf
  constexpr int LAK = KPAD + 8;
  __shared__ __align__(16) __bf16 As[BB * LAK];     // As[b][k]

  const int n   = blockIdx.x;
  const int tid = threadIdx.x;

  // Gather xs row block for this node: As[b][k], k = f*5 + m
  for (int idx = tid; idx < BB * KPAD; idx += 256) {
    int k = idx >> 5;
    int b = idx & 31;
    __bf16 v = f2bf(0.f);
    if (k < KTOT) {
      int f = k / MMAT, mm = k - f * MMAT;
      const __bf16* src = (mm == 0) ? X0 : (mm == 1) ? X1 : (mm == 2) ? X2
                          : (mm == 3) ? X3 : X4;
      v = src[(size_t)n * CC + f * BB + b];   // contiguous over b -> coalesced
    }
    As[b * LAK + k] = v;
  }
  __syncthreads();

  const int lane = tid & 31;
  const int wid  = tid >> 5;
  const int rowT = (wid & 1) * 16;                  // batch-tile (0 or 16)
  constexpr int CT = (OUT == 128) ? 2 : 1;          // col tiles per wave
  const int colBase = (wid >> 1) * (16 * CT);

  f32x8 acc[CT] = {};
  const int kh8  = (lane >> 4) << 3;
  const int kh16 = (lane >> 4) << 4;
  for (int kt = 0; kt < KPAD; kt += 32) {
    bf16x16 a;
    int m = rowT + (lane & 15);
#pragma unroll
    for (int i = 0; i < 8; ++i) {
      a[i]     = As[m * LAK + kt + kh8 + i];
      a[i + 8] = As[m * LAK + kt + 16 + kh8 + i];
    }
#pragma unroll
    for (int c = 0; c < CT; ++c) {
      bf16x16 b;
      int o = colBase + c * 16 + (lane & 15);
#pragma unroll
      for (int i = 0; i < 16; ++i) b[i] = Wt[(size_t)o * KPAD + kt + (i >> 3) * 16 + kh8 + (i & 7)];
      acc[c] = __builtin_amdgcn_wmma_f32_16x16x32_bf16(
          false, a, false, b, (short)0, acc[c], false, false);
    }
  }

#pragma unroll
  for (int c = 0; c < CT; ++c) {
    int o = colBase + c * 16 + (lane & 15);
#pragma unroll
    for (int r = 0; r < 8; ++r) {
      int b = rowT + r + ((lane >> 4) << 3);
      float v = acc[c][r] + bias[o];
      size_t base = (size_t)b * (NN * UU) + (size_t)n * UU;
      if constexpr (MODE == 0) {
        float g = 1.f / (1.f + expf(-v));
        if (o < UU) out0[base + o]      = g * hx[base + o];
        else        out1[base + o - UU] = g;
      } else {
        float cv = tanhf(v);
        size_t idx = base + o;
        float uu = ubuf[idx];
        out0[idx] = uu * hx[idx] + (1.f - uu) * cv;
      }
    }
  }
}

// ---------------------------------------------------------------- launcher
extern "C" void kernel_launch(void* const* d_in, const int* in_sizes, int n_in,
                              void* d_out, int out_size, void* d_ws, size_t ws_size,
                              hipStream_t stream) {
  const float* inputs = (const float*)d_in[0];
  const float* hx     = (const float*)d_in[1];
  const float* S0     = (const float*)d_in[2];
  const float* S1     = (const float*)d_in[3];
  const float* W1     = (const float*)d_in[4];
  const float* b1     = (const float*)d_in[5];
  const float* W2     = (const float*)d_in[6];
  const float* b2     = (const float*)d_in[7];
  float* out = (float*)d_out;

  char* p = (char*)d_ws;
  auto alloc = [&](size_t bytes) -> char* {
    char* q = p; p += (bytes + 255) & ~(size_t)255; return q;
  };
  __bf16* S0b = (__bf16*)alloc((size_t)NN * NN * 2);
  __bf16* S1b = (__bf16*)alloc((size_t)NN * NN * 2);
  __bf16* X[5];
  for (int m = 0; m < 5; ++m) X[m] = (__bf16*)alloc((size_t)NN * CC * 2);
  __bf16* Wt1 = (__bf16*)alloc((size_t)128 * KPAD * 2);
  __bf16* Wt2 = (__bf16*)alloc((size_t)64 * KPAD * 2);
  float* rhx  = (float*)alloc((size_t)BB * NN * UU * 4);
  float* ub   = (float*)alloc((size_t)BB * NN * UU * 4);

  // Demote supports to bf16 (L2-resident: 2 x 32 MB), pack weights.
  k_f32_to_bf16<<<(NN * NN + 255) / 256, 256, 0, stream>>>(S0, S0b, NN * NN);
  k_f32_to_bf16<<<(NN * NN + 255) / 256, 256, 0, stream>>>(S1, S1b, NN * NN);
  k_pack_w<<<(128 * KPAD + 255) / 256, 256, 0, stream>>>(W1, Wt1, 128);
  k_pack_w<<<(64 * KPAD + 255) / 256, 256, 0, stream>>>(W2, Wt2, 64);

  dim3 gg(NN / GT_M, CC / GT_N);   // 32 x 33
  int nbx = (int)(((size_t)NN * CC + 255) / 256);

  auto gconv = [&](const float* state) {
    k_build_x<<<nbx, 256, 0, stream>>>(inputs, state, X[0]);
    k_diff_gemm<false><<<gg, 256, 0, stream>>>(S0b, X[0], X[0], X[1], 1.f,  0.f);
    k_diff_gemm<true ><<<gg, 256, 0, stream>>>(S0b, X[1], X[0], X[2], 2.f, -1.f);
    k_diff_gemm<false><<<gg, 256, 0, stream>>>(S1b, X[0], X[0], X[3], 1.f,  0.f);
    k_diff_gemm<true ><<<gg, 256, 0, stream>>>(S1b, X[3], X[0], X[4], 2.f, -1.f);
  };

  // gconv 1: gates -> r*hx and u
  gconv(hx);
  k_proj<128, 0><<<NN, 256, 0, stream>>>(X[0], X[1], X[2], X[3], X[4],
                                         Wt1, b1, hx, nullptr, rhx, ub);
  // gconv 2: candidate -> fused final blend into d_out
  gconv(rhx);
  k_proj<64, 1><<<NN, 256, 0, stream>>>(X[0], X[1], X[2], X[3], X[4],
                                        Wt2, b2, hx, ub, out, nullptr);
  (void)in_sizes; (void)n_in; (void)out_size; (void)ws_size;
}
